// QuantizedLinearWhisper_635655160006
// MI455X (gfx1250) — compile-verified, matrix operand order checked by reading, not run on hardware
//
#include <hip/hip_runtime.h>
#include <hip/hip_bf16.h>

// ---------- types ----------
typedef _Float16 v8h  __attribute__((ext_vector_type(8)));
typedef _Float16 v16h __attribute__((ext_vector_type(16)));
typedef float    v8f  __attribute__((ext_vector_type(8)));

// ---------- problem constants ----------
#define DIM_M 12000      // 8*1500
#define DIM_N 5120
#define DIM_K 1280
#define NB_K  (DIM_K / 32)          // 40 K-blocks per row
#define NBLK_X (DIM_M * NB_K)       // 480000
#define NBLK_W (DIM_N * NB_K)       // 204800

// ============================================================
// E2M1 block-32 fake-quant: f32 -> dequantized f16 (+ scale)
// One thread handles one 32-element block (128B in, 64B out).
// ============================================================
__device__ __forceinline__ float e2m1_level(float mag) {
    float lvl = (mag > 0.25f) ? 0.5f : 0.0f;
    lvl = (mag > 0.75f) ? 1.0f : lvl;
    lvl = (mag > 1.25f) ? 1.5f : lvl;
    lvl = (mag > 1.75f) ? 2.0f : lvl;
    lvl = (mag > 2.5f)  ? 3.0f : lvl;
    lvl = (mag > 3.5f)  ? 4.0f : lvl;
    lvl = (mag > 5.0f)  ? 6.0f : lvl;
    return lvl;
}

__global__ void quant_e2m1_kernel(const float* __restrict__ in,
                                  _Float16* __restrict__ outq,
                                  float* __restrict__ scale_out,
                                  int nblocks) {
    int b = blockIdx.x * blockDim.x + threadIdx.x;
    if (b >= nblocks) return;

    const float4* p = (const float4*)(in + (size_t)b * 32);
    float4 v[8];
    float amax = 0.0f;
#pragma unroll
    for (int i = 0; i < 8; ++i) {
        v[i] = p[i];
        amax = fmaxf(amax, fabsf(v[i].x));
        amax = fmaxf(amax, fabsf(v[i].y));
        amax = fmaxf(amax, fabsf(v[i].z));
        amax = fmaxf(amax, fabsf(v[i].w));
    }
    float scale = fmaxf(amax * (1.0f / 6.0f), 1e-12f);
    float inv   = 1.0f / scale;

    _Float16 q[32];
#pragma unroll
    for (int i = 0; i < 8; ++i) {
        const float e[4] = {v[i].x, v[i].y, v[i].z, v[i].w};
#pragma unroll
        for (int j = 0; j < 4; ++j) {
            float x   = e[j];
            float lvl = e2m1_level(fabsf(x) * inv);
            q[i * 4 + j] = (_Float16)copysignf(lvl * scale, x);
        }
    }
    v8h* o = (v8h*)(outq + (size_t)b * 32);
#pragma unroll
    for (int i = 0; i < 4; ++i)
        o[i] = *(const v8h*)&q[i * 8];

    scale_out[b] = scale;
}

// ============================================================
// f16 WMMA GEMM: out[M,N] = Aq[M,K] * Wq[N,K]^T + bias[N]
// Wave tile 32(M) x 64(N) -> 8 accumulators, 8 WMMA / K-step,
// 12 b128 loads / K-step (1.5 loads per WMMA).
// Block: 8 waves arranged 1(M) x 8(N) -> block tile 32 x 512.
// Grid (375, 10): all dims divide exactly.
// Register double-buffering hides load latency behind WMMAs.
// ============================================================
__device__ __forceinline__ v16h load_frag(const _Float16* p) {
    v16h v;
    *(v8h*)&v       = *(const v8h*)(p);        // K = koff .. koff+7
    *((v8h*)&v + 1) = *(const v8h*)(p + 16);   // K = 16+koff .. 16+koff+7
    return v;
}

__global__ void __launch_bounds__(256)
wmma_gemm_f16_kernel(const _Float16* __restrict__ Aq,
                     const _Float16* __restrict__ Wq,
                     const float* __restrict__ bias,
                     float* __restrict__ out) {
    const int lane = threadIdx.x & 31;
    const int wave = threadIdx.x >> 5;          // 0..7  (N direction)

    const int mBase = blockIdx.x * 32;
    const int nBase = blockIdx.y * 512 + wave * 64;

    const int r    = lane & 15;                 // row (A) / col (B) within 16
    const int koff = (lane >> 4) * 8;           // K sub-offset (ISA 16-bit layout)

    const _Float16* aRow0 = Aq + (size_t)(mBase + r) * DIM_K + koff;
    const _Float16* aRow1 = aRow0 + (size_t)16 * DIM_K;
    const _Float16* bRow[4];
#pragma unroll
    for (int f = 0; f < 4; ++f)
        bRow[f] = Wq + (size_t)(nBase + f * 16 + r) * DIM_K + koff;

    v8f acc[2][4];
#pragma unroll
    for (int i = 0; i < 2; ++i)
#pragma unroll
        for (int f = 0; f < 4; ++f)
            acc[i][f] = v8f{};

    // ---- prologue: load K-step 0 ----
    v16h a0 = load_frag(aRow0);
    v16h a1 = load_frag(aRow1);
    v16h b0 = load_frag(bRow[0]);
    v16h b1 = load_frag(bRow[1]);
    v16h b2 = load_frag(bRow[2]);
    v16h b3 = load_frag(bRow[3]);

    for (int k0 = 0; k0 < DIM_K; k0 += 32) {
        // ---- prefetch next K-step into fresh registers ----
        // (clamped on the last iteration: redundant but branch-free)
        int kn = k0 + 32;
        kn = (kn < DIM_K) ? kn : k0;
        v16h na0 = load_frag(aRow0 + kn - k0 + k0);   // = aRow0 + kn
        na0 = load_frag(aRow0 + kn);
        v16h na1 = load_frag(aRow1 + kn);
        v16h nb0 = load_frag(bRow[0] + kn);
        v16h nb1 = load_frag(bRow[1] + kn);
        v16h nb2 = load_frag(bRow[2] + kn);
        v16h nb3 = load_frag(bRow[3] + kn);

        // ---- 8 back-to-back WMMAs on current fragments ----
        acc[0][0] = __builtin_amdgcn_wmma_f32_16x16x32_f16(false, a0, false, b0,
                    (short)0, acc[0][0], false, false);
        acc[0][1] = __builtin_amdgcn_wmma_f32_16x16x32_f16(false, a0, false, b1,
                    (short)0, acc[0][1], false, false);
        acc[0][2] = __builtin_amdgcn_wmma_f32_16x16x32_f16(false, a0, false, b2,
                    (short)0, acc[0][2], false, false);
        acc[0][3] = __builtin_amdgcn_wmma_f32_16x16x32_f16(false, a0, false, b3,
                    (short)0, acc[0][3], false, false);
        acc[1][0] = __builtin_amdgcn_wmma_f32_16x16x32_f16(false, a1, false, b0,
                    (short)0, acc[1][0], false, false);
        acc[1][1] = __builtin_amdgcn_wmma_f32_16x16x32_f16(false, a1, false, b1,
                    (short)0, acc[1][1], false, false);
        acc[1][2] = __builtin_amdgcn_wmma_f32_16x16x32_f16(false, a1, false, b2,
                    (short)0, acc[1][2], false, false);
        acc[1][3] = __builtin_amdgcn_wmma_f32_16x16x32_f16(false, a1, false, b3,
                    (short)0, acc[1][3], false, false);

        // ---- rotate buffers ----
        a0 = na0; a1 = na1;
        b0 = nb0; b1 = nb1; b2 = nb2; b3 = nb3;
    }

    // ---- epilogue: C/D layout ----
    // lane<16: N=lane, M=mTile+j ; lane>=16: N=lane-16, M=mTile+8+j
    const int col  = lane & 15;
    const int mOff = (lane >> 4) * 8;

    float bv[4];
#pragma unroll
    for (int f = 0; f < 4; ++f)
        bv[f] = bias[nBase + f * 16 + col];

#pragma unroll
    for (int i = 0; i < 2; ++i) {
#pragma unroll
        for (int j = 0; j < 8; ++j) {
            float* o = out + (size_t)(mBase + i * 16 + mOff + j) * DIM_N + nBase + col;
#pragma unroll
            for (int f = 0; f < 4; ++f)
                o[f * 16] = acc[i][f][j] + bv[f];
        }
    }
}

// ============================================================
// launch
// ============================================================
extern "C" void kernel_launch(void* const* d_in, const int* in_sizes, int n_in,
                              void* d_out, int out_size, void* d_ws, size_t ws_size,
                              hipStream_t stream) {
    const float* x      = (const float*)d_in[0];   // [8,1500,1280]
    const float* weight = (const float*)d_in[1];   // [5120,1280]
    const float* bias   = (const float*)d_in[2];   // [5120]

    float* out     = (float*)d_out;                         // [12000,5120]
    float* scale_w = (float*)d_out + (size_t)DIM_M * DIM_N; // [5120,40]

    // workspace layout
    char* ws = (char*)d_ws;
    _Float16* wq = (_Float16*)ws;                                   // 13.1 MB
    _Float16* xq = (_Float16*)(ws + (size_t)DIM_N * DIM_K * 2);     // 30.7 MB
    float* scale_x_dummy =
        (float*)(ws + (size_t)DIM_N * DIM_K * 2 + (size_t)DIM_M * DIM_K * 2); // 1.9 MB

    // 1) quantize weight (scales are a real output)
    quant_e2m1_kernel<<<(NBLK_W + 255) / 256, 256, 0, stream>>>(
        weight, wq, scale_w, NBLK_W);

    // 2) quantize activations (scales discarded into ws)
    quant_e2m1_kernel<<<(NBLK_X + 255) / 256, 256, 0, stream>>>(
        x, xq, scale_x_dummy, NBLK_X);

    // 3) f16 WMMA GEMM + bias
    dim3 grid(DIM_M / 32, DIM_N / 512);   // 375 x 10
    wmma_gemm_f16_kernel<<<grid, 256, 0, stream>>>(xq, wq, bias, out);
}
